// HiveGNNModel_16569983828059
// MI455X (gfx1250) — matrix operand order repaired and checked by reading, not compiled, not run on hardware
//
#include <hip/hip_runtime.h>
#include <math.h>

// ---------------- problem constants (match reference) ----------------
#define N_NODES 32768
#define F_IN    12
#define NHEAD   4
#define DHEAD   32
#define HIDF    128
#define NEDGE   524288
#define NGRAPH  64
#define MAXN    512
#define AOUT    512
#define NEG_SLOPE 0.2f
#define K0PAD   32            // layer-0 K padded to one WMMA K-step

typedef __bf16 bf16t;
typedef bf16t v16bf __attribute__((ext_vector_type(16)));
typedef float v8f   __attribute__((ext_vector_type(8)));
typedef unsigned int v4u __attribute__((ext_vector_type(4)));
typedef unsigned int u32x4 __attribute__((ext_vector_type(4)));
typedef int i32x8 __attribute__((ext_vector_type(8)));
typedef int i32x4 __attribute__((ext_vector_type(4)));

// ---------------- helpers ----------------
__device__ __forceinline__ float lrelu(float v) { return v > 0.f ? v : NEG_SLOPE * v; }

__device__ __forceinline__ void atomicMaxFloat(float* addr, float v) {
    // IEEE-ordered float max via integer atomics (works across mixed signs)
    if (v >= 0.f) atomicMax((int*)addr, __float_as_int(v));
    else          atomicMin((unsigned int*)addr, __float_as_uint(v));
}

// A-fragment: 16x32 bf16. lane<16: M=lane, K {0..7,16..23}; lane>=16: M=lane-16, K {8..15,24..31}
__device__ __forceinline__ v16bf load_a_frag(const bf16t* p) {
    union { v4u u[2]; v16bf v; } t;
    t.u[0] = *(const v4u*)(p);        // 8 bf16
    t.u[1] = *(const v4u*)(p + 16);   // 8 bf16
    return t.v;
}
// B-fragment: 32x16 bf16. lane<16: N=lane, K {0..7,16..23}; lane>=16: N=lane-16, K {8..15,24..31}
__device__ __forceinline__ v16bf load_b_frag(const bf16t* p) {
    union { v4u u[2]; v16bf v; } t;
    t.u[0] = *(const v4u*)(p);
    t.u[1] = *(const v4u*)(p + 16);
    return t.v;
}

// ---------------- generic bf16 WMMA GEMM: out[M][Ntot] = A[M][K] @ WT[Ntot][K]^T ----------------
// grid.x = M/32 (two 16-row tiles per wave, two independent accumulators),
// grid.y = Ntot/128, block = 256 (8 waves; wave w -> n-tile w)
__global__ __launch_bounds__(256)
void wmma_gemm_kernel(const bf16t* __restrict__ A, const bf16t* __restrict__ WT,
                      const float* __restrict__ bias, float* __restrict__ out,
                      int K, int Ntot, int relu) {
    extern __shared__ bf16t lds[];          // 128 * K bf16 weight slab
    const int lane = threadIdx.x & 31;
    const int wave = threadIdx.x >> 5;
    const int m0 = blockIdx.x * 32;
    const int c0 = blockIdx.y * 128;

    // ---- Tensor Data Mover: stage the 128-column weight slab (128*K bf16) into LDS.
    // One wave issues a single 1-D TDM descriptor; TENSORcnt tracks completion.
    if (threadIdx.x < 32) {
        unsigned lds_off = (unsigned)(unsigned long long)(const void*)lds; // flat[31:0] == LDS offset
        unsigned long long ga = (unsigned long long)(const void*)(WT + (size_t)c0 * K);
        unsigned nelem = 128u * (unsigned)K;                // elements (data_size = 2B)
        u32x4 g0 = {};
        g0[0] = 1u;                                        // count=1 user descriptor
        g0[1] = lds_off;                                   // lds_addr (bytes)
        g0[2] = (unsigned)(ga & 0xffffffffu);              // global_addr[31:0]
        g0[3] = (unsigned)((ga >> 32) & 0x1ffffffu) | 0x80000000u; // addr[56:32] | type=2
        i32x8 g1 = {};
        g1[0] = 1 << 16;                                   // data_size = 2 bytes
        g1[1] = (int)((nelem & 0xffffu) << 16);            // tensor_dim0[15:0]
        g1[2] = (int)((nelem >> 16) & 0xffffu) | (1 << 16);// tensor_dim0[31:16] | tensor_dim1=1
        g1[3] = (int)((nelem & 0xffffu) << 16);            // tile_dim0 = nelem (<=32768 fits u16)
        g1[4] = 0;                                         // tile_dim1/2 unused (1-D tile)
        g1[5] = (int)nelem;                                // tensor_dim0_stride[31:0]
        g1[6] = 0;                                         // stride hi / dim1_stride lo
        g1[7] = 0;
        i32x4 gz4 = {};
        i32x8 gz8 = {};
        __builtin_amdgcn_tensor_load_to_lds(g0, g1, gz4, gz4, gz8, 0);
        __builtin_amdgcn_s_wait_tensorcnt(0);
    }
    __syncthreads();

    v8f acc0 = {};
    v8f acc1 = {};
    const int r15 = lane & 15;
    const int kbA = (lane >> 4) << 3;       // 0 or 8
    const int kbB = (lane >> 4) << 3;       // 0 or 8
    const bf16t* arow0 = A   + (size_t)(m0 + r15) * K + kbA;
    const bf16t* arow1 = A   + (size_t)(m0 + 16 + r15) * K + kbA;
    const bf16t* brow  = lds + (size_t)(wave * 16 + r15) * K + kbB;

    for (int k0 = 0; k0 < K; k0 += 32) {
        __builtin_prefetch(arow0 + k0 + 256, 0, 0);  // global_prefetch_b8
        v16bf af0 = load_a_frag(arow0 + k0);
        v16bf af1 = load_a_frag(arow1 + k0);
        v16bf bw  = load_b_frag(brow + k0);
        // two independent accumulators -> no WMMA->WMMA RAW chain
        acc0 = __builtin_amdgcn_wmma_f32_16x16x32_bf16(
            false, af0, false, bw, (short)0, acc0, false, false);
        acc1 = __builtin_amdgcn_wmma_f32_16x16x32_bf16(
            false, af1, false, bw, (short)0, acc1, false, false);
    }

    const int col = c0 + wave * 16 + r15;
    const int mb  = m0 + ((lane >> 4) << 3);         // M 0..7 or 8..15 within tile
    const float bv = bias ? bias[col] : 0.0f;
#pragma unroll
    for (int g = 0; g < 8; ++g) {
        float v0 = acc0[g] + bv;
        float v1 = acc1[g] + bv;
        if (relu) { v0 = fmaxf(v0, 0.f); v1 = fmaxf(v1, 0.f); }
        out[(size_t)(mb + g) * Ntot + col]      = v0;
        out[(size_t)(mb + 16 + g) * Ntot + col] = v1;
    }
}

// ---------------- weight prep: WT[c][k] = (k<Ksrc ? W[k][c] : 0), bf16 ----------------
__global__ void prep_wt_kernel(const float* __restrict__ W, bf16t* __restrict__ WT,
                               int Ksrc, int Kpad, int Ncols) {
    int gid = blockIdx.x * blockDim.x + threadIdx.x;
    if (gid >= Ncols * Kpad) return;
    int c = gid / Kpad, k = gid - c * Kpad;
    WT[gid] = (k < Ksrc) ? (bf16t)W[(size_t)k * Ncols + c] : (bf16t)0.0f;
}

// ---------------- layer-0 input: pad F_IN=12 -> 32, convert to bf16 ----------------
__global__ void xconvert_kernel(const float* __restrict__ x, bf16t* __restrict__ hb) {
    int gid = blockIdx.x * blockDim.x + threadIdx.x;
    if (gid >= N_NODES * K0PAD) return;
    int n = gid >> 5, k = gid & 31;
    hb[gid] = (k < F_IN) ? (bf16t)x[n * F_IN + k] : (bf16t)0.0f;
}

// ---------------- per-layer init: out=bias, smax=-inf, den=0 ----------------
__global__ void layer_init_kernel(float* __restrict__ out, const float* __restrict__ bias,
                                  float* __restrict__ smax, float* __restrict__ den) {
    int gid = blockIdx.x * blockDim.x + threadIdx.x;
    if (gid < N_NODES * HIDF) out[gid] = bias[gid & (HIDF - 1)];
    if (gid < N_NODES * NHEAD) { smax[gid] = -INFINITY; den[gid] = 0.f; }
}

// ---------------- edge attention scores + segment max ----------------
__global__ void edge_scores_kernel(const float* __restrict__ xl, const float* __restrict__ xr,
                                   const int* __restrict__ src, const int* __restrict__ dst,
                                   const float* __restrict__ att,
                                   float* __restrict__ s, float* __restrict__ smax) {
    int gid = blockIdx.x * blockDim.x + threadIdx.x;
    if (gid >= NEDGE * NHEAD) return;
    int e = gid >> 2, h = gid & 3;
    int sn = src[e], dn = dst[e];
    const float4* pj = (const float4*)(xl + (size_t)sn * HIDF + h * DHEAD);
    const float4* pr = (const float4*)(xr + (size_t)dn * HIDF + h * DHEAD);
    const float4* pa = (const float4*)(att + h * DHEAD);
    float acc = 0.f;
#pragma unroll
    for (int i = 0; i < 8; ++i) {
        float4 a = pj[i], b = pr[i], w = pa[i];
        acc += w.x * lrelu(a.x + b.x) + w.y * lrelu(a.y + b.y)
             + w.z * lrelu(a.z + b.z) + w.w * lrelu(a.w + b.w);
    }
    s[gid] = acc;
    atomicMaxFloat(smax + (size_t)dn * NHEAD + h, acc);
}

// ---------------- exp(s - smax[dst]) and denominator ----------------
__global__ void edge_softmax_kernel(const int* __restrict__ dst, float* __restrict__ s,
                                    const float* __restrict__ smax, float* __restrict__ den) {
    int gid = blockIdx.x * blockDim.x + threadIdx.x;
    if (gid >= NEDGE * NHEAD) return;
    int e = gid >> 2, h = gid & 3;
    int dn = dst[e];
    float a = __expf(s[gid] - smax[(size_t)dn * NHEAD + h]);
    s[gid] = a;
    atomicAdd(den + (size_t)dn * NHEAD + h, a);
}

// ---------------- weighted scatter aggregation: out[dst] += a * xl[src] ----------------
__global__ void edge_aggregate_kernel(const float* __restrict__ s, const float* __restrict__ den,
                                      const float* __restrict__ xl, const int* __restrict__ src,
                                      const int* __restrict__ dst, float* __restrict__ out) {
    int gid = blockIdx.x * blockDim.x + threadIdx.x;   // NEDGE*32 threads, 4 floats each
    if (gid >= NEDGE * 32) return;
    int e = gid >> 5, q = gid & 31, h = q >> 3;
    int sn = src[e], dn = dst[e];
    float a = s[(size_t)e * NHEAD + h] / fmaxf(den[(size_t)dn * NHEAD + h], 1e-16f);
    float4 xj = *(const float4*)(xl + (size_t)sn * HIDF + q * 4);
    float* o = out + (size_t)dn * HIDF + q * 4;
    atomicAdd(o + 0, a * xj.x);
    atomicAdd(o + 1, a * xj.y);
    atomicAdd(o + 2, a * xj.z);
    atomicAdd(o + 3, a * xj.w);
}

// ---------------- ReLU in place + bf16 copy for next layer ----------------
__global__ void relu_conv_kernel(float* __restrict__ h, bf16t* __restrict__ hb) {
    int gid = blockIdx.x * blockDim.x + threadIdx.x;
    if (gid >= N_NODES * HIDF) return;
    float v = fmaxf(h[gid], 0.f);
    h[gid] = v;
    hb[gid] = (bf16t)v;
}

// ---------------- per-graph mean/max pooling (graphs are contiguous 512-node blocks) ----------------
__global__ void pool_kernel(const float* __restrict__ h, float* __restrict__ g,
                            bf16t* __restrict__ gb) {
    int b = blockIdx.x, f = threadIdx.x;   // 64 blocks x 128 threads
    const float* hp = h + (size_t)b * MAXN * HIDF + f;
    float sum = 0.f, mx = -INFINITY;
    for (int i = 0; i < MAXN; ++i) { float v = hp[(size_t)i * HIDF]; sum += v; mx = fmaxf(mx, v); }
    float mean = sum * (1.0f / MAXN);
    g [b * 2 * HIDF + f]        = mean;
    g [b * 2 * HIDF + HIDF + f] = mx;
    gb[b * 2 * HIDF + f]        = (bf16t)mean;
    gb[b * 2 * HIDF + HIDF + f] = (bf16t)mx;
}

// ---------------- f32 -> bf16 ----------------
__global__ void f2b_kernel(const float* __restrict__ in, bf16t* __restrict__ out, int n) {
    int gid = blockIdx.x * blockDim.x + threadIdx.x;
    if (gid < n) out[gid] = (bf16t)in[gid];
}

// ---------------- value head: val[b] = hv[b] . vW2 + vb2 ----------------
__global__ void val_kernel(const float* __restrict__ hv, const float* __restrict__ vW2,
                           const float* __restrict__ vb2, float* __restrict__ val) {
    int b = blockIdx.x * blockDim.x + threadIdx.x;
    if (b >= NGRAPH) return;
    float acc = vb2[0];
    for (int k = 0; k < HIDF; ++k) acc += hv[(size_t)b * HIDF + k] * vW2[k];
    val[b] = acc;
}

// ---------------- dueling combine: q = val + adv - mean(adv) ----------------
__global__ void head_final_kernel(const float* __restrict__ adv, const float* __restrict__ val,
                                  float* __restrict__ q) {
    __shared__ float red[256];
    int b = blockIdx.x, t = threadIdx.x;   // 256 threads, 2 elements each
    float v0 = adv[(size_t)b * AOUT + t];
    float v1 = adv[(size_t)b * AOUT + 256 + t];
    red[t] = v0 + v1;
    __syncthreads();
    for (int s = 128; s > 0; s >>= 1) {
        if (t < s) red[t] += red[t + s];
        __syncthreads();
    }
    float mean = red[0] * (1.0f / AOUT);
    float vb = val[b];
    q[(size_t)b * AOUT + t]       = vb + v0 - mean;
    q[(size_t)b * AOUT + 256 + t] = vb + v1 - mean;
}

// ================= host launcher =================
extern "C" void kernel_launch(void* const* d_in, const int* in_sizes, int n_in,
                              void* d_out, int out_size, void* d_ws, size_t ws_size,
                              hipStream_t stream) {
    (void)in_sizes; (void)n_in; (void)out_size; (void)ws_size;
    const float* x    = (const float*)d_in[0];
    const int*   esrc = (const int*)d_in[1];
    const int*   edst = (const int*)d_in[2];
    // d_in[3] = batch_idx (graphs are contiguous blocks; not needed)
    const float* Wl0  = (const float*)d_in[4];
    const float* Wr0  = (const float*)d_in[5];
    const float* att0 = (const float*)d_in[6];
    const float* b0   = (const float*)d_in[7];
    const float* Wl   = (const float*)d_in[8];
    const float* Wr   = (const float*)d_in[9];
    const float* att  = (const float*)d_in[10];
    const float* bb   = (const float*)d_in[11];
    const float* qW1  = (const float*)d_in[12];
    const float* qb1  = (const float*)d_in[13];
    const float* qW2  = (const float*)d_in[14];
    const float* qb2  = (const float*)d_in[15];
    const float* vW1  = (const float*)d_in[16];
    const float* vb1  = (const float*)d_in[17];
    const float* vW2  = (const float*)d_in[18];
    const float* vb2  = (const float*)d_in[19];
    float* q_out = (float*)d_out;

    char* ws = (char*)d_ws;
    size_t off = 0;
    auto alloc = [&](size_t bytes) -> void* {
        void* p = ws + off;
        off = (off + bytes + 255) & ~(size_t)255;
        return p;
    };

    float* xl    = (float*)alloc(sizeof(float) * (size_t)N_NODES * HIDF);
    float* xr    = (float*)alloc(sizeof(float) * (size_t)N_NODES * HIDF);
    float* hcur  = (float*)alloc(sizeof(float) * (size_t)N_NODES * HIDF);
    float* sbuf  = (float*)alloc(sizeof(float) * (size_t)NEDGE * NHEAD);
    float* smax  = (float*)alloc(sizeof(float) * (size_t)N_NODES * NHEAD);
    float* den   = (float*)alloc(sizeof(float) * (size_t)N_NODES * NHEAD);
    bf16t* hb    = (bf16t*)alloc(2 * (size_t)N_NODES * HIDF);
    float* gpool = (float*)alloc(sizeof(float) * NGRAPH * 2 * HIDF);
    bf16t* gb    = (bf16t*)alloc(2 * NGRAPH * 2 * HIDF);
    float* hq    = (float*)alloc(sizeof(float) * NGRAPH * HIDF);
    bf16t* hqb   = (bf16t*)alloc(2 * NGRAPH * HIDF);
    float* hv    = (float*)alloc(sizeof(float) * NGRAPH * HIDF);
    float* adv   = (float*)alloc(sizeof(float) * NGRAPH * AOUT);
    float* val   = (float*)alloc(sizeof(float) * NGRAPH);
    bf16t* wt0l  = (bf16t*)alloc(2 * 128 * K0PAD);
    bf16t* wt0r  = (bf16t*)alloc(2 * 128 * K0PAD);
    bf16t* wtl   = (bf16t*)alloc(2 * 3 * HIDF * HIDF);
    bf16t* wtr   = (bf16t*)alloc(2 * 3 * HIDF * HIDF);
    bf16t* wq1   = (bf16t*)alloc(2 * 128 * 256);
    bf16t* wq2   = (bf16t*)alloc(2 * 512 * 128);
    bf16t* wv1   = (bf16t*)alloc(2 * 128 * 256);

    const int TB = 256;
    auto nb = [](int n, int t) { return (n + t - 1) / t; };

    // weight prep (bf16 transposed)
    prep_wt_kernel<<<nb(128 * K0PAD, TB), TB, 0, stream>>>(Wl0, wt0l, F_IN, K0PAD, 128);
    prep_wt_kernel<<<nb(128 * K0PAD, TB), TB, 0, stream>>>(Wr0, wt0r, F_IN, K0PAD, 128);
    for (int i = 0; i < 3; ++i) {
        prep_wt_kernel<<<nb(HIDF * HIDF, TB), TB, 0, stream>>>(
            Wl + (size_t)i * HIDF * HIDF, wtl + (size_t)i * HIDF * HIDF, HIDF, HIDF, HIDF);
        prep_wt_kernel<<<nb(HIDF * HIDF, TB), TB, 0, stream>>>(
            Wr + (size_t)i * HIDF * HIDF, wtr + (size_t)i * HIDF * HIDF, HIDF, HIDF, HIDF);
    }
    prep_wt_kernel<<<nb(128 * 256, TB), TB, 0, stream>>>(qW1, wq1, 256, 256, 128);
    prep_wt_kernel<<<nb(512 * 128, TB), TB, 0, stream>>>(qW2, wq2, 128, 128, 512);
    prep_wt_kernel<<<nb(128 * 256, TB), TB, 0, stream>>>(vW1, wv1, 256, 256, 128);

    // layer-0 input
    xconvert_kernel<<<nb(N_NODES * K0PAD, TB), TB, 0, stream>>>(x, hb);

    // 4 GATv2 layers
    for (int layer = 0; layer < 4; ++layer) {
        const int K = (layer == 0) ? K0PAD : HIDF;
        const bf16t* wl_t  = (layer == 0) ? wt0l : wtl + (size_t)(layer - 1) * HIDF * HIDF;
        const bf16t* wr_t  = (layer == 0) ? wt0r : wtr + (size_t)(layer - 1) * HIDF * HIDF;
        const float* bias  = (layer == 0) ? b0   : bb  + (size_t)(layer - 1) * HIDF;
        const float* att_p = (layer == 0) ? att0 : att + (size_t)(layer - 1) * HIDF;
        const size_t ldsB = (size_t)128 * K * 2;

        wmma_gemm_kernel<<<dim3(N_NODES / 32, 1), 256, ldsB, stream>>>(hb, wl_t, nullptr, xl, K, HIDF, 0);
        wmma_gemm_kernel<<<dim3(N_NODES / 32, 1), 256, ldsB, stream>>>(hb, wr_t, nullptr, xr, K, HIDF, 0);
        layer_init_kernel<<<nb(N_NODES * HIDF, TB), TB, 0, stream>>>(hcur, bias, smax, den);
        edge_scores_kernel<<<nb(NEDGE * NHEAD, TB), TB, 0, stream>>>(xl, xr, esrc, edst, att_p, sbuf, smax);
        edge_softmax_kernel<<<nb(NEDGE * NHEAD, TB), TB, 0, stream>>>(edst, sbuf, smax, den);
        edge_aggregate_kernel<<<nb(NEDGE * 32, TB), TB, 0, stream>>>(sbuf, den, xl, esrc, edst, hcur);
        relu_conv_kernel<<<nb(N_NODES * HIDF, TB), TB, 0, stream>>>(hcur, hb);
    }

    // pooling + dueling head
    pool_kernel<<<NGRAPH, 128, 0, stream>>>(hcur, gpool, gb);

    wmma_gemm_kernel<<<dim3(NGRAPH / 32, 1), 256, (size_t)128 * 256 * 2, stream>>>(
        gb, wq1, qb1, hq, 256, HIDF, 1);
    f2b_kernel<<<nb(NGRAPH * HIDF, TB), TB, 0, stream>>>(hq, hqb, NGRAPH * HIDF);
    wmma_gemm_kernel<<<dim3(NGRAPH / 32, AOUT / 128), 256, (size_t)128 * 128 * 2, stream>>>(
        hqb, wq2, qb2, adv, HIDF, AOUT, 0);

    wmma_gemm_kernel<<<dim3(NGRAPH / 32, 1), 256, (size_t)128 * 256 * 2, stream>>>(
        gb, wv1, vb1, hv, 256, HIDF, 1);
    val_kernel<<<1, NGRAPH, 0, stream>>>(hv, vW2, vb2, val);

    head_final_kernel<<<NGRAPH, 256, 0, stream>>>(adv, val, q_out);
}
